// SwinTransformerBlock_13941463843247
// MI455X (gfx1250) — matrix operand order repaired and checked by reading, not compiled, not run on hardware
//
#include <hip/hip_runtime.h>
#include <math.h>

// ---------------------------------------------------------------------------
// Types for CDNA5 WMMA (wave32, v_wmma_f32_16x16x32_bf16)
// ---------------------------------------------------------------------------
typedef __bf16 bf16_t;
typedef __attribute__((ext_vector_type(16))) __bf16 v16bf;
typedef __attribute__((ext_vector_type(8)))  __bf16 v8bf;
typedef __attribute__((ext_vector_type(8)))  float  v8f;
typedef __attribute__((ext_vector_type(4)))  unsigned int u32x4;

union Frag16 { v16bf v; v8bf h[2]; };

#define TOK    100352          // B*H*W = 32*56*56
#define CDIM   384
#define NHEAD  12
#define HDIM   32
#define NWIN   2048            // 32 * 8 * 8 windows
#define QKVN   1152
#define MLPH   1536

// ---------------------------------------------------------------------------
// fp32 -> bf16 convert (weights), vectorized: b128 load / b64 store
// ---------------------------------------------------------------------------
__global__ void cvt_bf16_kernel(const float* __restrict__ src, bf16_t* __restrict__ dst, int n4)
{
    int i = blockIdx.x * blockDim.x + threadIdx.x;
    int stride = gridDim.x * blockDim.x;
    for (; i < n4; i += stride) {
        float4 f = ((const float4*)src)[i];
        union { uint2 u; bf16_t h[4]; } o;
        o.h[0] = (bf16_t)f.x; o.h[1] = (bf16_t)f.y;
        o.h[2] = (bf16_t)f.z; o.h[3] = (bf16_t)f.w;
        ((uint2*)dst)[i] = o.u;
    }
}

// ---------------------------------------------------------------------------
// LayerNorm (one wave per 384-wide row), fp32 in -> bf16 out
// ---------------------------------------------------------------------------
__global__ __launch_bounds__(256) void layernorm_kernel(
    const float* __restrict__ x, const float* __restrict__ g, const float* __restrict__ b,
    bf16_t* __restrict__ outp)
{
    const int lane = threadIdx.x & 31;
    const int wave = threadIdx.x >> 5;
    const int row  = blockIdx.x * 8 + wave;
    if (row >= TOK) return;
    const float4* xr = (const float4*)(x + (size_t)row * CDIM);
    float4 v[3];
    float s = 0.0f;
    #pragma unroll
    for (int i = 0; i < 3; ++i) {
        v[i] = xr[lane + i * 32];
        s += v[i].x + v[i].y + v[i].z + v[i].w;
    }
    #pragma unroll
    for (int off = 16; off >= 1; off >>= 1) s += __shfl_xor(s, off);
    float mean = s * (1.0f / 384.0f);
    float vs = 0.0f;
    #pragma unroll
    for (int i = 0; i < 3; ++i) {
        float d0 = v[i].x - mean, d1 = v[i].y - mean, d2 = v[i].z - mean, d3 = v[i].w - mean;
        vs += d0 * d0 + d1 * d1 + d2 * d2 + d3 * d3;
    }
    #pragma unroll
    for (int off = 16; off >= 1; off >>= 1) vs += __shfl_xor(vs, off);
    float rstd = rsqrtf(vs * (1.0f / 384.0f) + 1e-5f);
    #pragma unroll
    for (int i = 0; i < 3; ++i) {
        int c = (lane + i * 32) * 4;
        union { uint2 u; bf16_t h[4]; } o;
        o.h[0] = (bf16_t)((v[i].x - mean) * rstd * g[c + 0] + b[c + 0]);
        o.h[1] = (bf16_t)((v[i].y - mean) * rstd * g[c + 1] + b[c + 1]);
        o.h[2] = (bf16_t)((v[i].z - mean) * rstd * g[c + 2] + b[c + 2]);
        o.h[3] = (bf16_t)((v[i].w - mean) * rstd * g[c + 3] + b[c + 3]);
        *(uint2*)(outp + (size_t)row * CDIM + c) = o.u;
    }
}

// ---------------------------------------------------------------------------
// Generic bf16 GEMM + fused epilogue, double-buffered LDS (ping-pong).
//   C = A[M,KT] * B[KT,N] + bias (+resid) (+GELU), out bf16 or f32
//   Block: 256 thr = 8 waves; macro tile 128x128; wave tile 32x64; BK=32.
// ---------------------------------------------------------------------------
template<int KT, bool GELU, bool RESID, bool OUTF32>
__global__ __launch_bounds__(256) void gemm_bf16_kernel(
    const bf16_t* __restrict__ A, const bf16_t* __restrict__ Bw,
    const float* __restrict__ bias, const float* __restrict__ resid,
    void* __restrict__ outp, int M, int N)
{
    constexpr int LDT = 48;   // LDS row stride (elems): 96B, keeps 16B/32B alignment
    __shared__ __align__(64) bf16_t As[2][128 * LDT];
    __shared__ __align__(64) bf16_t Bs[2][128 * LDT];

    const int tid   = threadIdx.x;
    const int lane  = tid & 31;
    const int wave  = tid >> 5;
    const int hl    = lane & 15;
    const int koffA = (lane < 16) ? 0 : 16;   // A: K 0..15 vs 16..31 (contiguous)
    const int koffB = (lane < 16) ? 0 : 16;   // B: K 0..15 vs 16..31 (contiguous)
    const int rgrp  = (lane < 16) ? 0 : 8;    // C/D row group

    const int ntiles = N >> 7;
    const int m0 = (blockIdx.x / ntiles) << 7;
    const int n0 = (blockIdx.x % ntiles) << 7;
    const int wm = wave & 3;   // 4 waves along M (4*32 = 128)
    const int wn = wave >> 2;  // 2 waves along N (2*64 = 128)

    v8f acc[2][4];
    #pragma unroll
    for (int i = 0; i < 2; ++i)
        #pragma unroll
        for (int j = 0; j < 4; ++j)
            #pragma unroll
            for (int e = 0; e < 8; ++e) acc[i][j][e] = 0.0f;

    // register-staged tiles
    u32x4 ra[2];
    unsigned int rb[8];

    // preload k0 = 0
    #pragma unroll
    for (int r = 0; r < 2; ++r) {
        int i = tid + (r << 8);
        ra[r] = *(const u32x4*)(A + (size_t)(m0 + (i >> 2)) * KT + ((i & 3) << 3));
    }
    #pragma unroll
    for (int r = 0; r < 8; ++r) {
        int u = tid + (r << 8);
        rb[r] = *(const unsigned int*)(Bw + (size_t)(u >> 6) * N + n0 + ((u & 63) << 1));
    }

    int pb = 0;
    for (int k0 = 0; k0 < KT; k0 += 32) {
        // commit staged registers to LDS buffer pb
        #pragma unroll
        for (int r = 0; r < 2; ++r) {
            int i = tid + (r << 8);
            *(u32x4*)(&As[pb][(i >> 2) * LDT + ((i & 3) << 3)]) = ra[r];
        }
        #pragma unroll
        for (int r = 0; r < 8; ++r) {
            int u = tid + (r << 8);
            union { unsigned int u32; bf16_t h[2]; } cv; cv.u32 = rb[r];
            int cp = u & 63, kk = u >> 6;
            Bs[pb][(cp * 2)     * LDT + kk] = cv.h[0];
            Bs[pb][(cp * 2 + 1) * LDT + kk] = cv.h[1];
        }
        __syncthreads();

        // issue next tile's global loads before consuming LDS[pb]
        if (k0 + 32 < KT) {
            #pragma unroll
            for (int r = 0; r < 2; ++r) {
                int i = tid + (r << 8);
                const bf16_t* gp = A + (size_t)(m0 + (i >> 2)) * KT + k0 + 32 + ((i & 3) << 3);
                ra[r] = *(const u32x4*)gp;
                if (k0 + 64 < KT) __builtin_prefetch(gp + 32, 0, 1);
            }
            #pragma unroll
            for (int r = 0; r < 8; ++r) {
                int u = tid + (r << 8);
                const bf16_t* gp = Bw + (size_t)(k0 + 32 + (u >> 6)) * N + n0 + ((u & 63) << 1);
                rb[r] = *(const unsigned int*)gp;
                if (k0 + 64 < KT) __builtin_prefetch(gp + (size_t)32 * N, 0, 1);
            }
        }

        // compute from LDS[pb]
        Frag16 af[2];
        #pragma unroll
        for (int mt = 0; mt < 2; ++mt) {
            const bf16_t* p = &As[pb][(wm * 32 + mt * 16 + hl) * LDT];
            af[mt].h[0] = *(const v8bf*)(p + koffA);
            af[mt].h[1] = *(const v8bf*)(p + koffA + 8);
        }
        #pragma unroll
        for (int nt = 0; nt < 4; ++nt) {
            Frag16 bfr;
            bfr.v = *(const v16bf*)(&Bs[pb][(wn * 64 + nt * 16 + hl) * LDT + koffB]);
            #pragma unroll
            for (int mt = 0; mt < 2; ++mt) {
                acc[mt][nt] = __builtin_amdgcn_wmma_f32_16x16x32_bf16(
                    false, af[mt].v, false, bfr.v, (short)0, acc[mt][nt], false, false);
            }
        }
        pb ^= 1;
    }

    // --- epilogue: bias (+resid) (+GELU), store ---
    #pragma unroll
    for (int mt = 0; mt < 2; ++mt) {
        #pragma unroll
        for (int nt = 0; nt < 4; ++nt) {
            int col = n0 + wn * 64 + nt * 16 + hl;
            float bc = bias[col];
            #pragma unroll
            for (int e = 0; e < 8; ++e) {
                int row = m0 + wm * 32 + mt * 16 + rgrp + e;
                float v = acc[mt][nt][e] + bc;
                if (RESID) v += resid[(size_t)row * N + col];
                if (GELU)  v = 0.5f * v * (1.0f + erff(v * 0.70710678118654752f));
                if (OUTF32) ((float*)outp)[(size_t)row * N + col] = v;
                else        ((bf16_t*)outp)[(size_t)row * N + col] = (bf16_t)v;
            }
        }
    }
}

// ---------------------------------------------------------------------------
// Windowed attention: one block (4 waves, 128 thr) per (window, head).
// Shift + window partition handled as a token-index gather; output scattered
// back to flat token order so proj GEMM needs no reordering.
// Bias/mask pre-padded to 64x64 in LDS -> branch-free score fixup.
// ---------------------------------------------------------------------------
__global__ __launch_bounds__(128) void swin_attn_kernel(
    const bf16_t* __restrict__ qkv, const float* __restrict__ btab,
    const int* __restrict__ ridx, bf16_t* __restrict__ outp)
{
    __shared__ __align__(64) bf16_t Qs[64 * 32];   // padded 49->64 rows
    __shared__ __align__(64) bf16_t Ks[64 * 32];
    __shared__ __align__(64) bf16_t Vt[32 * 64];   // V transposed: [d][key]
    __shared__ __align__(64) bf16_t Ps[64 * 64];   // softmax probs (bf16)
    __shared__ __align__(64) float  Bb[64 * 64];   // bias + pad mask, padded
    __shared__ int Tok[64];

    const int tid  = threadIdx.x;
    const int lane = tid & 31;
    const int wave = tid >> 5;   // M-tile owner: rows [wave*16, wave*16+16)
    const int w    = blockIdx.x;
    const int head = blockIdx.y;

    // token gather indices (cyclic shift by +3 folded in)
    if (tid < 64) {
        int tok = -1;
        if (tid < 49) {
            int b = w >> 6, wi = w & 63;
            int wh = wi >> 3, ww = wi & 7;
            int r = tid / 7, c = tid % 7;
            int hh = (wh * 7 + r + 3) % 56;
            int cc = (ww * 7 + c + 3) % 56;
            tok = (b * 56 + hh) * 56 + cc;
        }
        Tok[tid] = tok;
    }
    // padded bias/mask table: valid -> rel bias, key-pad cols -> -1e30, else 0
    for (int i = tid; i < 64 * 64; i += 128) {
        int qr = i >> 6, ky = i & 63;
        float val;
        if (ky >= 49)              val = -1e30f;
        else if (qr < 49)          val = btab[ridx[qr * 49 + ky] * NHEAD + head];
        else                       val = 0.0f;
        Bb[i] = val;
    }
    __syncthreads();

    // stage Q (pre-scaled), K, V^T
    {
        int row = tid >> 1;
        int hc  = (tid & 1) << 4;
        int tok = Tok[row];
        if (tok >= 0) {
            const bf16_t* base = qkv + (size_t)tok * QKVN + head * HDIM + hc;
            #pragma unroll
            for (int j = 0; j < 16; ++j) {
                Qs[row * 32 + hc + j] = (bf16_t)((float)base[j] * 0.17677669529663687f);
                Ks[row * 32 + hc + j] = base[384 + j];
                Vt[(hc + j) * 64 + row] = base[768 + j];
            }
        } else {
            #pragma unroll
            for (int j = 0; j < 16; ++j) {
                Qs[row * 32 + hc + j] = (bf16_t)0.0f;
                Ks[row * 32 + hc + j] = (bf16_t)0.0f;
                Vt[(hc + j) * 64 + row] = (bf16_t)0.0f;
            }
        }
    }
    __syncthreads();

    const int hl    = lane & 15;
    const int koffA = (lane < 16) ? 0 : 16;
    const int koffB = (lane < 16) ? 0 : 16;
    const int rgrp  = (lane < 16) ? 0 : 8;

    // S = Q * K^T  (one K-step of 32 = HDIM)
    Frag16 aq;
    {
        const bf16_t* p = Qs + (wave * 16 + hl) * 32;
        aq.h[0] = *(const v8bf*)(p + koffA);
        aq.h[1] = *(const v8bf*)(p + koffA + 8);
    }
    v8f sa[4];
    #pragma unroll
    for (int nt = 0; nt < 4; ++nt) {
        Frag16 bk;
        bk.v = *(const v16bf*)(Ks + (nt * 16 + hl) * 32 + koffB);
        v8f z;
        #pragma unroll
        for (int e = 0; e < 8; ++e) z[e] = 0.0f;
        sa[nt] = __builtin_amdgcn_wmma_f32_16x16x32_bf16(
            false, aq.v, false, bk.v, (short)0, z, false, false);
    }

    // branch-free bias + mask from padded table
    #pragma unroll
    for (int nt = 0; nt < 4; ++nt) {
        int ky = nt * 16 + hl;
        #pragma unroll
        for (int e = 0; e < 8; ++e) {
            int qr = wave * 16 + rgrp + e;
            sa[nt][e] += Bb[qr * 64 + ky];
        }
    }

    // softmax along keys: reduce over 4 tiles + 16-lane shfl groups
    #pragma unroll
    for (int e = 0; e < 8; ++e) {
        float m = fmaxf(fmaxf(sa[0][e], sa[1][e]), fmaxf(sa[2][e], sa[3][e]));
        #pragma unroll
        for (int off = 1; off < 16; off <<= 1) m = fmaxf(m, __shfl_xor(m, off));
        float p[4];
        float sum = 0.0f;
        #pragma unroll
        for (int nt = 0; nt < 4; ++nt) { p[nt] = __expf(sa[nt][e] - m); sum += p[nt]; }
        #pragma unroll
        for (int off = 1; off < 16; off <<= 1) sum += __shfl_xor(sum, off);
        float inv = 1.0f / sum;
        int qr = wave * 16 + rgrp + e;
        #pragma unroll
        for (int nt = 0; nt < 4; ++nt)
            Ps[qr * 64 + nt * 16 + hl] = (bf16_t)(p[nt] * inv);
    }
    // each wave reads back only the P rows it wrote -> no block barrier needed

    // O = P * V  (K = 64 -> 2 steps)
    v8f oa[2];
    #pragma unroll
    for (int nt = 0; nt < 2; ++nt)
        #pragma unroll
        for (int e = 0; e < 8; ++e) oa[nt][e] = 0.0f;
    #pragma unroll
    for (int ks = 0; ks < 2; ++ks) {
        Frag16 ap;
        const bf16_t* pp = Ps + (wave * 16 + hl) * 64 + ks * 32;
        ap.h[0] = *(const v8bf*)(pp + koffA);
        ap.h[1] = *(const v8bf*)(pp + koffA + 8);
        #pragma unroll
        for (int nt = 0; nt < 2; ++nt) {
            Frag16 bv;
            bv.v = *(const v16bf*)(Vt + (nt * 16 + hl) * 64 + ks * 32 + koffB);
            oa[nt] = __builtin_amdgcn_wmma_f32_16x16x32_bf16(
                false, ap.v, false, bv.v, (short)0, oa[nt], false, false);
        }
    }

    // scatter O back to flat token order (window reverse + unshift)
    #pragma unroll
    for (int nt = 0; nt < 2; ++nt) {
        #pragma unroll
        for (int e = 0; e < 8; ++e) {
            int qr = wave * 16 + rgrp + e;
            if (qr < 49) {
                int tok = Tok[qr];
                outp[(size_t)tok * CDIM + head * HDIM + nt * 16 + hl] = (bf16_t)oa[nt][e];
            }
        }
    }
}

// ---------------------------------------------------------------------------
// Host-side orchestration
// ---------------------------------------------------------------------------
extern "C" void kernel_launch(void* const* d_in, const int* in_sizes, int n_in,
                              void* d_out, int out_size, void* d_ws, size_t ws_size,
                              hipStream_t stream)
{
    const float* x      = (const float*)d_in[0];
    const float* n1g    = (const float*)d_in[1];
    const float* n1b    = (const float*)d_in[2];
    const float* qkv_w  = (const float*)d_in[3];
    const float* qkv_b  = (const float*)d_in[4];
    const float* proj_w = (const float*)d_in[5];
    const float* proj_b = (const float*)d_in[6];
    const float* relt   = (const float*)d_in[7];
    const float* n2g    = (const float*)d_in[8];
    const float* n2b    = (const float*)d_in[9];
    const float* fc1_w  = (const float*)d_in[10];
    const float* fc1_b  = (const float*)d_in[11];
    const float* fc2_w  = (const float*)d_in[12];
    const float* fc2_b  = (const float*)d_in[13];
    const int*   relidx = (const int*)d_in[14];

    char* ws = (char*)d_ws;
    size_t off = 0;
    auto alloc = [&](size_t bytes) -> char* {
        char* p = ws + off;
        off += (bytes + 255) & ~(size_t)255;
        return p;
    };
    bf16_t* buf0  = (bf16_t*)alloc((size_t)TOK * CDIM * 2);  // LN out / attn out / LN2 out
    bf16_t* bufQA = (bf16_t*)alloc((size_t)TOK * MLPH * 2);  // qkv (1152) then fc1 act (1536)
    float*  bufH  = (float*) alloc((size_t)TOK * CDIM * 4);  // residual h (fp32)
    bf16_t* wqkv  = (bf16_t*)alloc((size_t)CDIM * QKVN * 2);
    bf16_t* wproj = (bf16_t*)alloc((size_t)CDIM * CDIM * 2);
    bf16_t* wfc1  = (bf16_t*)alloc((size_t)CDIM * MLPH * 2);
    bf16_t* wfc2  = (bf16_t*)alloc((size_t)MLPH * CDIM * 2);

    // 1) weight conversion fp32 -> bf16 (vectorized x4)
    cvt_bf16_kernel<<<512, 256, 0, stream>>>(qkv_w,  wqkv,  CDIM * QKVN / 4);
    cvt_bf16_kernel<<<512, 256, 0, stream>>>(proj_w, wproj, CDIM * CDIM / 4);
    cvt_bf16_kernel<<<512, 256, 0, stream>>>(fc1_w,  wfc1,  CDIM * MLPH / 4);
    cvt_bf16_kernel<<<512, 256, 0, stream>>>(fc2_w,  wfc2,  MLPH * CDIM / 4);

    // 2) LN1(x) -> buf0 (bf16)
    layernorm_kernel<<<TOK / 8, 256, 0, stream>>>(x, n1g, n1b, buf0);

    // 3) QKV GEMM: [TOK,384] x [384,1152] -> bufQA (bf16)
    gemm_bf16_kernel<384, false, false, false><<<(TOK / 128) * (QKVN / 128), 256, 0, stream>>>(
        buf0, wqkv, qkv_b, nullptr, bufQA, TOK, QKVN);

    // 4) windowed attention -> buf0 (bf16, flat token order)
    swin_attn_kernel<<<dim3(NWIN, NHEAD), 128, 0, stream>>>(bufQA, relt, relidx, buf0);

    // 5) proj GEMM + residual(x): -> bufH (fp32)
    gemm_bf16_kernel<384, false, true, true><<<(TOK / 128) * (CDIM / 128), 256, 0, stream>>>(
        buf0, wproj, proj_b, x, bufH, TOK, CDIM);

    // 6) LN2(h) -> buf0 (bf16)
    layernorm_kernel<<<TOK / 8, 256, 0, stream>>>(bufH, n2g, n2b, buf0);

    // 7) FC1 GEMM + exact GELU: [TOK,384] x [384,1536] -> bufQA (bf16)
    gemm_bf16_kernel<384, true, false, false><<<(TOK / 128) * (MLPH / 128), 256, 0, stream>>>(
        buf0, wfc1, fc1_b, nullptr, bufQA, TOK, MLPH);

    // 8) FC2 GEMM + residual(h): [TOK,1536] x [1536,384] -> d_out (fp32)
    gemm_bf16_kernel<1536, false, true, true><<<(TOK / 128) * (CDIM / 128), 256, 0, stream>>>(
        bufQA, wfc2, fc2_b, bufH, d_out, TOK, CDIM);
}
